// NeuralCDE_76390288327550
// MI455X (gfx1250) — matrix operand order repaired and checked by reading, not compiled, not run on hardware
//
#include <hip/hip_runtime.h>
#include <hip/hip_bf16.h>
#include <math.h>

// ---------------------------------------------------------------------------
// Neural CDE (RK4 over linear-interp control path), persistent-WG design.
//  B=512, T=512, IN=6, H=64, HH=128, OUT=8
//  - 32 workgroups, each owns 16 batch rows for the whole time scan.
//  - All MLP weights live in LDS as BF16, N-major (pre-transposed), padded
//    rows (+8 elems) for bank-conflict-free half-wave loads.
//  - GEMMs via v_wmma_f32_16x16x32_bf16, f32 accumulate; relu/tanh/RK4 in f32.
//  - Weight LDS fill via GLOBAL_LOAD_ASYNC_TO_LDS_B128 when available.
// ---------------------------------------------------------------------------

typedef __bf16 bf16_t;
typedef bf16_t v16bf __attribute__((ext_vector_type(16)));
typedef bf16_t v8bf  __attribute__((ext_vector_type(8)));
typedef float  v8f   __attribute__((ext_vector_type(8)));
typedef int    v4i   __attribute__((ext_vector_type(4)));
typedef __attribute__((address_space(1))) v4i* gv4i_p;   // global int4*
typedef __attribute__((address_space(3))) v4i* lv4i_p;   // LDS int4*

#define B_TOT 512
#define T_TOT 512
#define IN_D  6
#define H_D   64
#define HH_D  128
#define OUT_D 8

#define BT       16            // batch rows per workgroup
#define NWAVES   8
#define NTHREADS (NWAVES * 32)

// Padded row strides (bf16 elements) for bank-conflict-free LDS access.
#define S64   72               // 64 + 8
#define S128  136              // 128 + 8

// Packed-weight (d_ws) / LDS bf16 layout (element offsets)
#define OFF_WIN   0                        // 128 x S64   (W_in^T)
#define OFF_WH1   (OFF_WIN + 128 * S64)    // 128 x S128  (W_h1^T)
#define OFF_WH2   (OFF_WH1 + 128 * S128)   // 128 x S128  (W_h2^T)
#define OFF_WOUT  (OFF_WH2 + 128 * S128)   // 384 x S128  (W_out^T)
#define NPACK     (OFF_WOUT + 384 * S128)  // 96256 bf16 elements
#define OFF_HIN   NPACK                    // 16 x S64  bf16 activations
#define OFF_Z1    (OFF_HIN + BT * S64)     // 16 x S128
#define OFF_Z2    (OFF_Z1 + BT * S128)     // 16 x S128
#define NBF16     (OFF_Z2 + BT * S128)     // total bf16 elems in LDS

// f32 LDS region (float offsets)
#define F_H     0
#define F_HACC  (F_H + BT * H_D)
#define F_A     (F_HACC + BT * H_D)           // 16 x 384 tanh output
#define F_DX    (F_A + BT * H_D * IN_D)       // 16 x 8 (padded from 6)
#define F_BIN   (F_DX + BT * 8)
#define F_BH1   (F_BIN + HH_D)
#define F_BH2   (F_BH1 + HH_D)
#define F_BOUT  (F_BH2 + HH_D)
#define F_BINIT (F_BOUT + H_D * IN_D)
#define F_WINIT (F_BINIT + H_D)               // 6 x 64
#define F_WFIN  (F_WINIT + IN_D * H_D)        // 64 x 8
#define F_BFIN  (F_WFIN + H_D * OUT_D)
#define NF32    (F_BFIN + 8)

#define SMEM_BYTES (NBF16 * 2 + NF32 * 4)

// ---------------------------------------------------------------------------
// Prologue: convert fp32 weights -> bf16, transposed to N-major, padded rows.
// ---------------------------------------------------------------------------
__global__ void ncde_pack_weights(const float* __restrict__ W_in,
                                  const float* __restrict__ W_h1,
                                  const float* __restrict__ W_h2,
                                  const float* __restrict__ W_out,
                                  bf16_t* __restrict__ ws) {
  for (int i = blockIdx.x * blockDim.x + threadIdx.x; i < NPACK;
       i += gridDim.x * blockDim.x) {
    float v = 0.0f;
    if (i < OFF_WH1) {                       // W_in (64,128) -> [n][k], k<64
      int n = i / S64, k = i % S64;
      if (k < 64) v = W_in[k * 128 + n];
    } else if (i < OFF_WH2) {                // W_h1 (128,128)
      int j = i - OFF_WH1, n = j / S128, k = j % S128;
      if (k < 128) v = W_h1[k * 128 + n];
    } else if (i < OFF_WOUT) {               // W_h2 (128,128)
      int j = i - OFF_WH2, n = j / S128, k = j % S128;
      if (k < 128) v = W_h2[k * 128 + n];
    } else {                                 // W_out (128,384)
      int j = i - OFF_WOUT, n = j / S128, k = j % S128;
      if (k < 128) v = W_out[k * 384 + n];
    }
    ws[i] = (bf16_t)v;
  }
}

// ---------------------------------------------------------------------------
// One MLP layer: out[16 x NT*16] = act( A[16 x K] @ W^T + bias )
//   A   : LDS bf16, row stride AS
//   WT  : LDS bf16, N-major, row stride WS (row n = column n of W, K values)
//   MODE 0: relu -> bf16 LDS (row stride OS) ; MODE 1: tanh -> f32 LDS
//   Tile loop is a compile-time rep loop (NT % NWAVES == 0) so the per-wave
//   work is uniform: no divergent branch / saveexec around the WMMA blocks.
// ---------------------------------------------------------------------------
template <int K, int NT, int MODE, int AS, int WS, int OS>
__device__ __forceinline__ void layer(const bf16_t* __restrict__ Ain,
                                      const bf16_t* __restrict__ WT,
                                      const float* __restrict__ bias,
                                      bf16_t* __restrict__ outB,
                                      float* __restrict__ outF) {
  static_assert(NT % NWAVES == 0, "tiles must divide evenly across waves");
  const int wid  = threadIdx.x >> 5;
  const int lane = threadIdx.x & 31;
  const int col  = lane & 15;        // A row / B column within tile
  const int hi   = lane >> 4;        // half-wave select
  const int aoff = hi * 8;           // A: lanes 16-31 hold K 8-15 / 24-31
  const int boff = hi * 8;           // B: lanes 16-31 hold K 8-15 / 24-31

#pragma unroll
  for (int rep = 0; rep < NT / NWAVES; ++rep) {
    const int nt = rep * NWAVES + wid;
    const int n = nt * 16 + col;
    v8f acc = {};
#pragma unroll
    for (int k0 = 0; k0 < K; k0 += 32) {
      union { v16bf v; v8bf h[2]; } a, b;
      const bf16_t* ap = Ain + col * AS + k0 + aoff;
      a.h[0] = *(const v8bf*)(ap);           // K = k0+{0..7} (+8 if hi)
      a.h[1] = *(const v8bf*)(ap + 16);      // K = k0+{16..23} (+8 if hi)
      const bf16_t* bp = WT + n * WS + k0 + boff;
      b.h[0] = *(const v8bf*)(bp);           // K = k0+{0..7} (+8 if hi)
      b.h[1] = *(const v8bf*)(bp + 16);      // K = k0+{16..23} (+8 if hi)
      acc = __builtin_amdgcn_wmma_f32_16x16x32_bf16(
          false, a.v, false, b.v, (short)0, acc, false, false);
    }
    const float bv = bias[n];
#pragma unroll
    for (int r = 0; r < 8; ++r) {
      const int m = r + hi * 8;              // output row (batch)
      float v = acc[r] + bv;
      if (MODE == 0) {
        v = v > 0.0f ? v : 0.0f;
        outB[m * OS + n] = (bf16_t)v;
      } else {
        outF[m * OS + n] = tanhf(v);
      }
    }
  }
}

// ---------------------------------------------------------------------------
// einsum("bhi,bi->bh") with A=tanh(...) plus RK4 stage combine.
// ---------------------------------------------------------------------------
__device__ __forceinline__ void rk4_update(int stage,
                                           const float* __restrict__ Af,
                                           const float* __restrict__ dXs,
                                           float* __restrict__ hF,
                                           float* __restrict__ haccF,
                                           bf16_t* __restrict__ hinB) {
  for (int p = threadIdx.x; p < BT * H_D; p += NTHREADS) {
    const int b = p >> 6, hd = p & 63;
    const float* ar = Af + b * (H_D * IN_D) + hd * IN_D;
    const float* dx = dXs + b * 8;
    float k = 0.0f;
#pragma unroll
    for (int i = 0; i < IN_D; ++i) k += ar[i] * dx[i];
    const float h0 = hF[p];
    const int hp = b * S64 + hd;
    if (stage == 0) {
      haccF[p] = h0 + k * (1.0f / 6.0f);
      hinB[hp] = (bf16_t)(h0 + 0.5f * k);
    } else if (stage == 1) {
      haccF[p] += k * (1.0f / 3.0f);
      hinB[hp] = (bf16_t)(h0 + 0.5f * k);
    } else if (stage == 2) {
      haccF[p] += k * (1.0f / 3.0f);
      hinB[hp] = (bf16_t)(h0 + k);
    } else {
      const float hn = haccF[p] + k * (1.0f / 6.0f);
      hF[p] = hn;
      hinB[hp] = (bf16_t)hn;                 // stage-0 input of next step
    }
  }
}

// ---------------------------------------------------------------------------
// Persistent kernel: one workgroup = 16 batch rows for all 511 RK4 steps.
// ---------------------------------------------------------------------------
__global__ __launch_bounds__(NTHREADS, 1)
void ncde_kernel(const float* __restrict__ x,
                 const float* __restrict__ W_init, const float* __restrict__ b_init,
                 const float* __restrict__ b_in,  const float* __restrict__ b_h1,
                 const float* __restrict__ b_h2,  const float* __restrict__ b_out,
                 const float* __restrict__ W_fin, const float* __restrict__ b_fin,
                 const bf16_t* __restrict__ wpack,
                 float* __restrict__ out) {
  extern __shared__ char smem[];
  bf16_t* sb   = (bf16_t*)smem;
  bf16_t* wtin = sb + OFF_WIN;
  bf16_t* wth1 = sb + OFF_WH1;
  bf16_t* wth2 = sb + OFF_WH2;
  bf16_t* wtout= sb + OFF_WOUT;
  bf16_t* hinB = sb + OFF_HIN;
  bf16_t* z1   = sb + OFF_Z1;
  bf16_t* z2   = sb + OFF_Z2;
  float* sf    = (float*)(smem + NBF16 * 2);
  float* hF    = sf + F_H;
  float* haccF = sf + F_HACC;
  float* Af    = sf + F_A;
  float* dXs   = sf + F_DX;
  float* sbin  = sf + F_BIN;
  float* sbh1  = sf + F_BH1;
  float* sbh2  = sf + F_BH2;
  float* sbout = sf + F_BOUT;
  float* sbinit= sf + F_BINIT;
  float* swinit= sf + F_WINIT;
  float* swfin = sf + F_WFIN;
  float* sbfin = sf + F_BFIN;

  const int tid = threadIdx.x;

  // ---- one-time: weights into LDS (async DMA path when available) --------
#if __has_builtin(__builtin_amdgcn_global_load_async_to_lds_b128)
  {
    const int bytes = NPACK * 2;             // 192512, 16B-aligned chunks
    for (int off = tid * 16; off < bytes; off += NTHREADS * 16) {
      __builtin_amdgcn_global_load_async_to_lds_b128(
          (gv4i_p)((char*)wpack + off),      // global src (int4, AS1)
          (lv4i_p)((char*)sb + off),         // LDS dst (int4, AS3)
          0, 0);
    }
#if __has_builtin(__builtin_amdgcn_s_wait_asynccnt)
    __builtin_amdgcn_s_wait_asynccnt(0);
#else
    asm volatile("s_wait_asynccnt 0x0" ::: "memory");
#endif
  }
#else
  for (int i = tid; i < NPACK; i += NTHREADS) sb[i] = wpack[i];
#endif
  for (int i = tid; i < HH_D; i += NTHREADS) {
    sbin[i] = b_in[i]; sbh1[i] = b_h1[i]; sbh2[i] = b_h2[i];
  }
  for (int i = tid; i < H_D * IN_D; i += NTHREADS) {
    sbout[i] = b_out[i]; swinit[i] = W_init[i];
  }
  for (int i = tid; i < H_D * OUT_D; i += NTHREADS) swfin[i] = W_fin[i];
  for (int i = tid; i < H_D; i += NTHREADS) sbinit[i] = b_init[i];
  if (tid < OUT_D) sbfin[tid] = b_fin[tid];
  __syncthreads();

  const int bg = blockIdx.x * BT;

  // ---- h0 = x[:,0] @ W_init + b_init -------------------------------------
  for (int p = tid; p < BT * H_D; p += NTHREADS) {
    const int b = p >> 6, hd = p & 63;
    const float* xr = x + (size_t)(bg + b) * T_TOT * IN_D;
    float acc = sbinit[hd];
#pragma unroll
    for (int i = 0; i < IN_D; ++i) acc += xr[i] * swinit[i * H_D + hd];
    hF[p] = acc;
    hinB[b * S64 + hd] = (bf16_t)acc;
  }
  __syncthreads();

  // ---- time scan ----------------------------------------------------------
  for (int t = 0; t < T_TOT - 1; ++t) {
    // emit out[:, t] from current h, and load dX for this interval
    if (tid < BT * OUT_D) {
      const int b = tid >> 3, o = tid & 7;
      const float* hr = hF + b * H_D;
      float acc = sbfin[o];
#pragma unroll 8
      for (int k = 0; k < H_D; ++k) acc += hr[k] * swfin[k * OUT_D + o];
      out[((size_t)(bg + b) * T_TOT + t) * OUT_D + o] = acc;
    }
    if (tid < BT * IN_D) {
      const int b = tid / IN_D, i = tid % IN_D;
      const float* xr = x + ((size_t)(bg + b) * T_TOT + t) * IN_D;
      dXs[b * 8 + i] = xr[IN_D + i] - xr[i];
      if (t + 2 < T_TOT) __builtin_prefetch(xr + 2 * IN_D, 0, 0);
    }
    __syncthreads();

#pragma unroll
    for (int s = 0; s < 4; ++s) {
      layer<64, 8, 0, S64, S64, S128>(hinB, wtin, sbin, z1, nullptr);
      __syncthreads();
      layer<128, 8, 0, S128, S128, S128>(z1, wth1, sbh1, z2, nullptr);
      __syncthreads();
      layer<128, 8, 0, S128, S128, S128>(z2, wth2, sbh2, z1, nullptr);
      __syncthreads();
      layer<128, 24, 1, S128, S128, H_D * IN_D>(z1, wtout, sbout, nullptr, Af);
      __syncthreads();
      rk4_update(s, Af, dXs, hF, haccF, hinB);
      __syncthreads();
    }
  }

  // ---- final time step output --------------------------------------------
  if (tid < BT * OUT_D) {
    const int b = tid >> 3, o = tid & 7;
    const float* hr = hF + b * H_D;
    float acc = sbfin[o];
#pragma unroll 8
    for (int k = 0; k < H_D; ++k) acc += hr[k] * swfin[k * OUT_D + o];
    out[((size_t)(bg + b) * T_TOT + (T_TOT - 1)) * OUT_D + o] = acc;
  }
}

// ---------------------------------------------------------------------------
extern "C" void kernel_launch(void* const* d_in, const int* in_sizes, int n_in,
                              void* d_out, int out_size, void* d_ws, size_t ws_size,
                              hipStream_t stream) {
  (void)in_sizes; (void)n_in; (void)out_size; (void)ws_size;
  const float* x      = (const float*)d_in[0];
  const float* W_init = (const float*)d_in[1];
  const float* b_init = (const float*)d_in[2];
  const float* W_in   = (const float*)d_in[3];
  const float* b_in   = (const float*)d_in[4];
  const float* W_h1   = (const float*)d_in[5];
  const float* b_h1   = (const float*)d_in[6];
  const float* W_h2   = (const float*)d_in[7];
  const float* b_h2   = (const float*)d_in[8];
  const float* W_out  = (const float*)d_in[9];
  const float* b_out  = (const float*)d_in[10];
  const float* W_fin  = (const float*)d_in[11];
  const float* b_fin  = (const float*)d_in[12];
  bf16_t* wpack = (bf16_t*)d_ws;

  ncde_pack_weights<<<96, 256, 0, stream>>>(W_in, W_h1, W_h2, W_out, wpack);
  ncde_kernel<<<B_TOT / BT, NTHREADS, SMEM_BYTES, stream>>>(
      x, W_init, b_init, b_in, b_h1, b_h2, b_out, W_fin, b_fin, wpack,
      (float*)d_out);
}
